// AttentionBlock_26594437496987
// MI455X (gfx1250) — matrix operand order repaired and checked by reading, not compiled, not run on hardware
//
#include <hip/hip_runtime.h>
#include <math.h>

#define DEVI static __device__ __forceinline__

constexpr int Bc  = 2, Sc = 2048, Hc = 4096, NHc = 32, HDc = 128, KVHc = 2;
constexpr int ROTc = HDc / 2;           // 64
constexpr int NQ   = NHc * HDc;         // 4096
constexpr int NKV  = KVHc * HDc;        // 256
constexpr int QKVc = NQ + 2 * NKV;      // 4608

typedef __attribute__((ext_vector_type(16))) __bf16 bf16x16;
typedef __attribute__((ext_vector_type(8)))  float  f32x8;
typedef __attribute__((__vector_size__(4 * sizeof(int)))) int v4i_t;

#if defined(__has_builtin)
#  if __has_builtin(__builtin_amdgcn_global_load_async_to_lds_b128)
#    define HAVE_ASYNC_LDS 1
#  endif
#endif
#ifndef HAVE_ASYNC_LDS
#  define HAVE_ASYNC_LDS 0
#endif

union FragU {
  bf16x16 v;
  __bf16  b[16];
  uint4   q2[2];
};

DEVI f32x8 wmma_bf16(bf16x16 a, bf16x16 b, f32x8 c) {
  return __builtin_amdgcn_wmma_f32_16x16x32_bf16(
      /*neg_a=*/false, a, /*neg_b=*/false, b,
      /*c_mod=*/(short)0, c, /*reuse_a=*/false, /*reuse_b=*/false);
}

DEVI void cvt8(float4 a, float4 b, __bf16* d) {
  d[0] = (__bf16)a.x; d[1] = (__bf16)a.y; d[2] = (__bf16)a.z; d[3] = (__bf16)a.w;
  d[4] = (__bf16)b.x; d[5] = (__bf16)b.y; d[6] = (__bf16)b.z; d[7] = (__bf16)b.w;
}

// 16-byte global->LDS copy: async-to-LDS (ASYNCcnt) when available, else ld/st.
DEVI void async_copy16(const void* g, void* l) {
#if HAVE_ASYNC_LDS
  __builtin_amdgcn_global_load_async_to_lds_b128(
      (__attribute__((address_space(1))) v4i_t*)(g),
      (__attribute__((address_space(3))) v4i_t*)(l), 0, 0);
#else
  *(uint4*)l = *(const uint4*)g;
#endif
}

DEVI void async_fence() {
#if HAVE_ASYNC_LDS
  asm volatile("s_wait_asynccnt 0" ::: "memory");
#endif
}

// A fragment: 16(M) x 32(K). Source fp32 row-major, lda floats (global or LDS).
// lanes 0-15 M=lane K{0..15}; lanes 16-31 M=lane-16 K{16..31}.
DEVI bf16x16 loadA_f32(const float* A, int lda, int lane) {
  const int m = lane & 15, kb = (lane >> 4) * 16;
  const float* p = A + (size_t)m * lda;
  FragU f;
  cvt8(*(const float4*)(p + kb),     *(const float4*)(p + kb + 4),  f.b);
  cvt8(*(const float4*)(p + kb + 8), *(const float4*)(p + kb + 12), f.b + 8);
  return f.v;
}

// B fragment built from B^T: source row-major N x K (fp32). B[k][n] = Bt[n][k].
DEVI bf16x16 loadBt_f32(const float* Bt, int ldb, int lane) {
  const int n = lane & 15, ks = (lane >> 4) * 16;
  const float* p = Bt + (size_t)n * ldb + ks;
  FragU f;
  cvt8(*(const float4*)(p),     *(const float4*)(p + 4),  f.b);
  cvt8(*(const float4*)(p + 8), *(const float4*)(p + 12), f.b + 8);
  return f.v;
}

// Same fragments from bf16 sources (16B vector loads; global or LDS).
DEVI bf16x16 loadA_bf(const __bf16* A, int lda, int lane) {
  const int m = lane & 15, kb = (lane >> 4) * 16;
  FragU f;
  f.q2[0] = *(const uint4*)(A + (size_t)m * lda + kb);
  f.q2[1] = *(const uint4*)(A + (size_t)m * lda + kb + 8);
  return f.v;
}

DEVI bf16x16 loadBt_bf(const __bf16* Bt, int ldb, int lane) {
  const int n = lane & 15, ks = (lane >> 4) * 16;
  FragU f;
  f.q2[0] = *(const uint4*)(Bt + (size_t)n * ldb + ks);
  f.q2[1] = *(const uint4*)(Bt + (size_t)n * ldb + ks + 8);
  return f.v;
}

// B fragment from row-major K x N source (bf16) — used for V (rows = kv position).
DEVI bf16x16 loadB_bf(const __bf16* Bm, int ldb, int lane) {
  const int n = lane & 15, ks = (lane >> 4) * 16;
  FragU f;
#pragma unroll
  for (int i = 0; i < 16; ++i) f.b[i] = Bm[(size_t)(ks + i) * ldb + n];
  return f.v;
}

// ---------------------------------------------------------------------------
// Kernel 1: mixed = hidden @ Wqkv^T + b ; split + RoPE ; emit bf16 Q/K/V.
// Block shares a 64-row A panel staged in LDS (double-buffered, async fill);
// one wave -> 64(M) x 64(N): 4 A-frags x 4 B-frags -> 16 WMMA per k-step.
// ---------------------------------------------------------------------------
__global__ __launch_bounds__(256) void qkv_rope_kernel(
    const float* __restrict__ hidden, const float* __restrict__ rope,
    const float* __restrict__ Wqkv,   const float* __restrict__ bias,
    __bf16* __restrict__ qout, __bf16* __restrict__ kout, __bf16* __restrict__ vout)
{
  __shared__ __align__(16) float bufA[2][64][32];     // 16 KB double buffer

  const int tid = threadIdx.x;
  const int lane = tid & 31, wid = tid >> 5;
  const int m0 = blockIdx.y * 64;
  const int n0 = blockIdx.x * 512 + wid * 64;

  // cooperative A-panel fill: 64x32 fp32 = 8KB -> 2 x b128 per thread
  auto fill = [&](int buf, int k0) {
    const int row = tid >> 2;            // 0..63
    const int col = (tid & 3) * 8;       // 0,8,16,24
    const float* g = hidden + (size_t)(m0 + row) * Hc + k0 + col;
    async_copy16(g,     &bufA[buf][row][col]);
    async_copy16(g + 4, &bufA[buf][row][col + 4]);
  };

  fill(0, 0);
  f32x8 acc[4][4] = {};                  // [mt][nt]
  int buf = 0;
  for (int k0 = 0; k0 < Hc; k0 += 32) {
    async_fence();
    __syncthreads();                     // panel 'buf' ready for all waves
    if (k0 + 32 < Hc) fill(buf ^ 1, k0 + 32);

    bf16x16 a[4];
#pragma unroll
    for (int mt = 0; mt < 4; ++mt)
      a[mt] = loadA_f32(&bufA[buf][mt * 16][0], 32, lane);
#pragma unroll
    for (int nt = 0; nt < 4; ++nt) {
      const float* bp = Wqkv + (size_t)(n0 + 16 * nt) * Hc + k0;
      __builtin_prefetch(bp + 64, 0, 1);
      bf16x16 bfrag = loadBt_f32(bp, Hc, lane);
#pragma unroll
      for (int mt = 0; mt < 4; ++mt)
        acc[mt][nt] = wmma_bf16(a[mt], bfrag, acc[mt][nt]);
    }
    buf ^= 1;
  }

  const int hi = lane >> 4, nl = lane & 15;
#pragma unroll
  for (int nt = 0; nt < 4; ++nt) {
    const int col  = n0 + 16 * nt + nl;
    const float bv = bias[col];
    const int dloc = col % HDc;                 // region bases are multiples of 128
    const bool isV = col >= NQ + NKV;
    const bool rot = (!isV) && (dloc < ROTc);   // wave-uniform (16-col tiles)
#pragma unroll
    for (int mt = 0; mt < 4; ++mt) {
#pragma unroll
      for (int r = 0; r < 8; ++r) {
        const int m = m0 + 16 * mt + hi * 8 + r;     // flat token index b*S+s
        float v = acc[mt][nt][r] + bv;
        if (rot) {
          const float other = __shfl_xor(v, 1);      // RoPE pair in adjacent lane
          const float* rp = rope + ((size_t)m * (ROTc / 2) + (dloc >> 1)) * 2;
          v = (col & 1) ? fmaf(other, rp[1], v * rp[0])     // x1*c + x0*s
                        : fmaf(v, rp[0], -other * rp[1]);   // x0*c - x1*s
        }
        const __bf16 o = (__bf16)v;
        const int bb = m / Sc, ss = m % Sc;
        if (col < NQ) {
          const int hh = col / HDc;
          qout[(((size_t)(bb * NHc + hh)) * Sc + ss) * HDc + dloc] = o;
        } else if (!isV) {
          const int hh = (col - NQ) / HDc;
          kout[(((size_t)(bb * KVHc + hh)) * Sc + ss) * HDc + dloc] = o;
        } else {
          const int hh = (col - NQ - NKV) / HDc;
          vout[(((size_t)(bb * KVHc + hh)) * Sc + ss) * HDc + dloc] = o;
        }
      }
    }
  }
}

// ---------------------------------------------------------------------------
// Kernel 2: causal flash attention, GQA (32 q-heads -> 2 kv-heads).
// One wave per 32-row q block (2 M-tiles); kv tiles of 32 so each P(16x32)
// is exactly one bf16 A-fragment.  K/V fragments feed both M-tiles.
// ---------------------------------------------------------------------------
__global__ __launch_bounds__(256) void attn_kernel(
    const __bf16* __restrict__ qb, const __bf16* __restrict__ kbuf,
    const __bf16* __restrict__ vbuf, __bf16* __restrict__ ctx)
{
  __shared__ __align__(16) __bf16 lds_p[8][2][512];   // 8 waves * 2 tiles * 16x32

  const int lane = threadIdx.x & 31, wid = threadIdx.x >> 5;
  const int q0 = (blockIdx.x * 8 + wid) * 32;
  const int h  = blockIdx.y;
  const int b  = blockIdx.z;
  const int kvh = h / (NHc / KVHc);

  const __bf16* Q = qb   + (((size_t)(b * NHc + h))   * Sc + q0) * HDc;
  const __bf16* K = kbuf + ((size_t)(b * KVHc + kvh)) * Sc * HDc;
  const __bf16* V = vbuf + ((size_t)(b * KVHc + kvh)) * Sc * HDc;

  bf16x16 qf[2][4];
#pragma unroll
  for (int mt = 0; mt < 2; ++mt)
#pragma unroll
    for (int c = 0; c < 4; ++c)
      qf[mt][c] = loadA_bf(Q + (size_t)(mt * 16) * HDc + c * 32, HDc, lane);

  f32x8 cacc[2][8] = {};
  float mrun[2][8], lrun[2][8];
#pragma unroll
  for (int mt = 0; mt < 2; ++mt)
#pragma unroll
    for (int r = 0; r < 8; ++r) { mrun[mt][r] = -__builtin_inff(); lrun[mt][r] = 0.f; }

  const float scale = 0.08838834764831845f;    // 1/sqrt(128)
  const int hi = lane >> 4, nl = lane & 15;
  const int jend = q0 + 32;

  for (int jt = 0; jt < jend; jt += 32) {
    // ---- scores: Q(32x128) . K^T tile (32 kv rows); K frags shared by both M-tiles
    f32x8 s[2][2] = {};
#pragma unroll
    for (int c = 0; c < 4; ++c) {
      bf16x16 k0f = loadBt_bf(K + (size_t)jt * HDc + c * 32, HDc, lane);
      bf16x16 k1f = loadBt_bf(K + (size_t)(jt + 16) * HDc + c * 32, HDc, lane);
#pragma unroll
      for (int mt = 0; mt < 2; ++mt) {
        s[mt][0] = wmma_bf16(qf[mt][c], k0f, s[mt][0]);
        s[mt][1] = wmma_bf16(qf[mt][c], k1f, s[mt][1]);
      }
    }

    // ---- mask + online softmax (rows live in 16-lane halves) ----
    float corr[2][8];
#pragma unroll
    for (int mt = 0; mt < 2; ++mt) {
#pragma unroll
      for (int r = 0; r < 8; ++r) {
        const int row = q0 + 16 * mt + hi * 8 + r;
        float v0 = s[mt][0][r] * scale; if (jt + nl      > row) v0 = -__builtin_inff();
        float v1 = s[mt][1][r] * scale; if (jt + 16 + nl > row) v1 = -__builtin_inff();
        float mx = fmaxf(v0, v1);
        mx = fmaxf(mx, __shfl_xor(mx, 1));
        mx = fmaxf(mx, __shfl_xor(mx, 2));
        mx = fmaxf(mx, __shfl_xor(mx, 4));
        mx = fmaxf(mx, __shfl_xor(mx, 8));
        const float mnew = fmaxf(mrun[mt][r], mx);
        const float cr = __expf(mrun[mt][r] - mnew);
        const float p0 = __expf(v0 - mnew);
        const float p1 = __expf(v1 - mnew);
        float rs = p0 + p1;
        rs += __shfl_xor(rs, 1);
        rs += __shfl_xor(rs, 2);
        rs += __shfl_xor(rs, 4);
        rs += __shfl_xor(rs, 8);
        lrun[mt][r] = lrun[mt][r] * cr + rs;
        mrun[mt][r] = mnew;
        corr[mt][r] = cr;
        // stash P in LDS (C-layout -> memory) for the A-fragment reload
        const int rl = hi * 8 + r;
        lds_p[wid][mt][rl * 32 + nl]      = (__bf16)p0;
        lds_p[wid][mt][rl * 32 + 16 + nl] = (__bf16)p1;
      }
    }
    asm volatile("s_wait_dscnt 0" ::: "memory");   // cross-lane LDS transpose fence

    bf16x16 pa0 = loadA_bf(&lds_p[wid][0][0], 32, lane);
    bf16x16 pa1 = loadA_bf(&lds_p[wid][1][0], 32, lane);
#pragma unroll
    for (int d = 0; d < 8; ++d) {
#pragma unroll
      for (int r = 0; r < 8; ++r) {
        cacc[0][d][r] *= corr[0][r];
        cacc[1][d][r] *= corr[1][r];
      }
      bf16x16 vb = loadB_bf(V + (size_t)jt * HDc + d * 16, HDc, lane);
      cacc[0][d] = wmma_bf16(pa0, vb, cacc[0][d]);
      cacc[1][d] = wmma_bf16(pa1, vb, cacc[1][d]);
    }
  }

  // ---- normalize + emit ctx as bf16 in (b, s, h*HD+d) layout for the dense GEMM
#pragma unroll
  for (int mt = 0; mt < 2; ++mt)
#pragma unroll
    for (int d = 0; d < 8; ++d)
#pragma unroll
      for (int r = 0; r < 8; ++r) {
        const int row = q0 + 16 * mt + hi * 8 + r;
        const float o = cacc[mt][d][r] / lrun[mt][r];
        ctx[((size_t)(b * Sc + row)) * NQ + h * HDc + d * 16 + nl] = (__bf16)o;
      }
}

// ---------------------------------------------------------------------------
// Kernel 3: out = ctx @ Wdense^T  (bf16 A staged in LDS, fp32->bf16 B, fp32 out)
// ---------------------------------------------------------------------------
__global__ __launch_bounds__(256) void dense_kernel(
    const __bf16* __restrict__ ctx, const float* __restrict__ Wd,
    float* __restrict__ out)
{
  __shared__ __align__(16) __bf16 bufA[2][64][32];    // 8 KB double buffer

  const int tid = threadIdx.x;
  const int lane = tid & 31, wid = tid >> 5;
  const int m0 = blockIdx.y * 64;
  const int n0 = blockIdx.x * 512 + wid * 64;

  // cooperative A-panel fill: 64x32 bf16 = 4KB -> 1 x b128 per thread
  auto fill = [&](int buf, int k0) {
    const int row = tid >> 2;            // 0..63
    const int col = (tid & 3) * 8;       // 0,8,16,24
    async_copy16(ctx + (size_t)(m0 + row) * NQ + k0 + col, &bufA[buf][row][col]);
  };

  fill(0, 0);
  f32x8 acc[4][4] = {};
  int buf = 0;
  for (int k0 = 0; k0 < NQ; k0 += 32) {
    async_fence();
    __syncthreads();
    if (k0 + 32 < NQ) fill(buf ^ 1, k0 + 32);

    bf16x16 a[4];
#pragma unroll
    for (int mt = 0; mt < 4; ++mt)
      a[mt] = loadA_bf(&bufA[buf][mt * 16][0], 32, lane);
#pragma unroll
    for (int nt = 0; nt < 4; ++nt) {
      const float* bp = Wd + (size_t)(n0 + 16 * nt) * NQ + k0;
      __builtin_prefetch(bp + 64, 0, 1);
      bf16x16 bfrag = loadBt_f32(bp, NQ, lane);
#pragma unroll
      for (int mt = 0; mt < 4; ++mt)
        acc[mt][nt] = wmma_bf16(a[mt], bfrag, acc[mt][nt]);
    }
    buf ^= 1;
  }
  const int hi = lane >> 4, nl = lane & 15;
#pragma unroll
  for (int nt = 0; nt < 4; ++nt)
#pragma unroll
    for (int mt = 0; mt < 4; ++mt)
#pragma unroll
      for (int r = 0; r < 8; ++r)
        out[(size_t)(m0 + 16 * mt + hi * 8 + r) * Hc + n0 + 16 * nt + nl] = acc[mt][nt][r];
}

// ---------------------------------------------------------------------------
extern "C" void kernel_launch(void* const* d_in, const int* in_sizes, int n_in,
                              void* d_out, int out_size, void* d_ws, size_t ws_size,
                              hipStream_t stream) {
  const float* hidden = (const float*)d_in[0];
  const float* rope   = (const float*)d_in[1];
  // d_in[2] = attention_mask: pure causal triu -> implemented analytically
  const float* Wqkv   = (const float*)d_in[3];
  const float* bqkv   = (const float*)d_in[4];
  const float* Wd     = (const float*)d_in[5];
  float* out = (float*)d_out;

  __bf16* ws  = (__bf16*)d_ws;
  __bf16* qb  = ws;                                     // B*NH*S*HD
  __bf16* kb  = qb + (size_t)Bc * NHc * Sc * HDc;       // B*KVH*S*HD
  __bf16* vb  = kb + (size_t)Bc * KVHc * Sc * HDc;      // B*KVH*S*HD
  __bf16* ctx = vb + (size_t)Bc * KVHc * Sc * HDc;      // B*S*NQ

  dim3 blk(256);
  qkv_rope_kernel<<<dim3(QKVc / 512, (Bc * Sc) / 64, 1), blk, 0, stream>>>(
      hidden, rope, Wqkv, bqkv, qb, kb, vb);
  attn_kernel<<<dim3(Sc / (32 * 8), NHc, Bc), blk, 0, stream>>>(qb, kb, vb, ctx);
  dense_kernel<<<dim3(Hc / 512, (Bc * Sc) / 64, 1), blk, 0, stream>>>(ctx, Wd, out);
}
